// CrossAttention_61735859912918
// MI455X (gfx1250) — compile-verified, matrix operand order checked by reading, not run on hardware
//
#include <hip/hip_runtime.h>
#include <hip/hip_bf16.h>
#include <stdint.h>

// Problem constants (from the reference)
#define B_   4
#define SQ_  2048
#define SKV_ 2048
#define D_   768
#define H_   12
#define HD_  16
#define DR_  192   // H*HD

typedef __attribute__((ext_vector_type(16))) __bf16 v16bf;
typedef __attribute__((ext_vector_type(8)))  float  v8f;
typedef int v4i_ __attribute__((vector_size(16)));

#define GAS_ __attribute__((address_space(1)))
#define LAS_ __attribute__((address_space(3)))

#if defined(__HIP_DEVICE_COMPILE__) && \
    __has_builtin(__builtin_amdgcn_global_load_async_to_lds_b128) && \
    __has_builtin(__builtin_amdgcn_s_wait_asynccnt)
#define USE_ASYNC_LDS 1
#else
#define USE_ASYNC_LDS 0
#endif

union FragAB {
    v16bf v;
    uint4 q[2];
    unsigned short u[16];
};

__device__ __forceinline__ unsigned short f2bf(float f) {
    union { float f; uint32_t u; } x; x.f = f;
    uint32_t r = x.u + 0x7FFFu + ((x.u >> 16) & 1u);   // round-to-nearest-even
    return (unsigned short)(r >> 16);
}

__device__ __forceinline__ float fast_exp2(float x) {
#if defined(__HIP_DEVICE_COMPILE__)
    return __builtin_amdgcn_exp2f(x);     // native v_exp_f32
#else
    return 0.0f;
#endif
}

// ---------------------------------------------------------------------------
// Tiled GEMM with bias: C(MxN) = A(MxK) * W(KxN) + bias, bf16 WMMA math.
// AT = float (convert on load) or ushort (already bf16). OT = float or ushort.
// Block: 256 threads = 8 waves; tile 128x64, BK=32; wave tile 32x32 (2x2 WMMA).
// ---------------------------------------------------------------------------
template <typename AT, typename OT>
__global__ __launch_bounds__(256) void gemm_bias_wmma_k(
    const AT* __restrict__ A, const float* __restrict__ W,
    const float* __restrict__ bias, OT* __restrict__ C,
    int M, int N, int K)
{
    __shared__ unsigned short As[128][40];  // row-major, padded (conflict-free)
    __shared__ unsigned short Bs[64][40];   // transposed W tile: Bs[n][k]

    const int t    = threadIdx.x;
    const int lane = t & 31;
    const int w    = t >> 5;
    const int wm   = w & 3;         // 4 waves along M
    const int wn   = w >> 2;        // 2 waves along N
    const int m0   = blockIdx.y * 128;
    const int n0   = blockIdx.x * 64;
    const int kh   = lane >> 4;     // lane half
    const int ln   = lane & 15;
    const int d0   = kh * 8;        // fragment k-offset for this half

    v8f acc[2][2];
    #pragma unroll
    for (int i = 0; i < 2; ++i)
        #pragma unroll
        for (int j = 0; j < 2; ++j)
            acc[i][j] = {};

    for (int kb = 0; kb < K; kb += 32) {
        // ---- stage A tile 128x32 into LDS (bf16) ----
        if constexpr (sizeof(AT) == 4) {
            #pragma unroll
            for (int it = 0; it < 4; ++it) {
                int gid = t + it * 256;                  // 1024 float4s
                int r = gid >> 3;
                int c = (gid & 7) * 4;
                const float4 vs = *(const float4*)(A + (size_t)(m0 + r) * K + kb + c);
                uint2 pk;
                pk.x = (uint32_t)f2bf(vs.x) | ((uint32_t)f2bf(vs.y) << 16);
                pk.y = (uint32_t)f2bf(vs.z) | ((uint32_t)f2bf(vs.w) << 16);
                *(uint2*)&As[r][c] = pk;
            }
        } else {
            #pragma unroll
            for (int it = 0; it < 2; ++it) {
                int gid = t + it * 256;                  // 512 uint4s (8 bf16 each)
                int r = gid >> 2;
                int c = (gid & 3) * 8;
                const unsigned short* gp = (const unsigned short*)A +
                                           (size_t)(m0 + r) * K + kb + c;
#if USE_ASYNC_LDS
                __builtin_amdgcn_global_load_async_to_lds_b128(
                    (GAS_ v4i_*)gp, (LAS_ v4i_*)&As[r][c], 0, 0);
#else
                *(uint4*)&As[r][c] = *(const uint4*)gp;
#endif
            }
        }
        // ---- stage W tile 32x64, transposed, into LDS (bf16) ----
        #pragma unroll
        for (int it = 0; it < 2; ++it) {
            int gid = t + it * 256;                      // 512 float4s
            int kr = gid >> 4;
            int c  = (gid & 15) * 4;
            const float4 vs = *(const float4*)(W + (size_t)(kb + kr) * N + n0 + c);
            Bs[c + 0][kr] = f2bf(vs.x);
            Bs[c + 1][kr] = f2bf(vs.y);
            Bs[c + 2][kr] = f2bf(vs.z);
            Bs[c + 3][kr] = f2bf(vs.w);
        }
#if USE_ASYNC_LDS
        if constexpr (sizeof(AT) == 2) __builtin_amdgcn_s_wait_asynccnt(0);
#endif
        __syncthreads();

        // ---- fragments + 4 WMMAs ----
        FragAB a[2], bfr[2];
        #pragma unroll
        for (int mf = 0; mf < 2; ++mf) {
            int r = wm * 32 + mf * 16 + ln;
            a[mf].q[0] = *(const uint4*)&As[r][d0];
            a[mf].q[1] = *(const uint4*)&As[r][d0 + 16];
        }
        #pragma unroll
        for (int nf = 0; nf < 2; ++nf) {
            int cidx = wn * 32 + nf * 16 + ln;
            bfr[nf].q[0] = *(const uint4*)&Bs[cidx][d0];
            bfr[nf].q[1] = *(const uint4*)&Bs[cidx][d0 + 16];
        }
        #pragma unroll
        for (int mf = 0; mf < 2; ++mf)
            #pragma unroll
            for (int nf = 0; nf < 2; ++nf)
                acc[mf][nf] = __builtin_amdgcn_wmma_f32_16x16x32_bf16(
                    false, a[mf].v, false, bfr[nf].v, (short)0, acc[mf][nf],
                    false, false);
        __syncthreads();
    }

    // ---- epilogue: + bias, store ----
    #pragma unroll
    for (int mf = 0; mf < 2; ++mf) {
        #pragma unroll
        for (int nf = 0; nf < 2; ++nf) {
            int n = n0 + wn * 32 + nf * 16 + ln;
            float bv = bias[n];
            #pragma unroll
            for (int i = 0; i < 8; ++i) {
                int r = m0 + wm * 32 + mf * 16 + i + 8 * kh;
                float val = acc[mf][nf][i] + bv;
                if constexpr (sizeof(OT) == 4)
                    ((float*)C)[(size_t)r * N + n] = val;
                else
                    ((unsigned short*)C)[(size_t)r * N + n] = f2bf(val);
            }
        }
    }
}

// ---------------------------------------------------------------------------
// Flash attention, head-dim 16. Block = 8 waves, each wave owns 16 q rows;
// block shares one (b,h) and streams KV in 64-row chunks through LDS.
// Softmax in exp2 domain; row sums computed by WMMA against a ones matrix
// (P @ 1) so no shuffle reduction is needed for the denominator.
// ---------------------------------------------------------------------------
__global__ __launch_bounds__(256) void attn_wmma_k(
    const unsigned short* __restrict__ Qb,   // (B*SQ, DR) bf16
    const unsigned short* __restrict__ KVb,  // (B*SKV, 2*DR) bf16; V at +DR
    unsigned short* __restrict__ Ob)         // (B*SQ, DR) bf16
{
    __shared__ unsigned short Ks[64][24];    // K chunk,  row-major (kv, d)
    __shared__ unsigned short Vt[16][72];    // V chunk,  transposed (d, kv)
    __shared__ unsigned short Ps[8][16][72]; // per-wave P buffer (16 x 64)

    const int t    = threadIdx.x;
    const int lane = t & 31;
    const int w    = t >> 5;
    const int kh   = lane >> 4;
    const int ln   = lane & 15;
    const int d0   = kh * 8;

    const int b    = blockIdx.z;
    const int h    = blockIdx.y;
    const int qblk = blockIdx.x;

    // Q A-fragment: row = lane%16 of this wave's 16-row tile; K-dim 16, padded to 32.
    const int qrow = b * SQ_ + qblk * 128 + w * 16 + ln;
    FragAB aq;
    aq.q[0] = *(const uint4*)(Qb + (size_t)qrow * DR_ + h * HD_ + d0);
    aq.q[1] = make_uint4(0u, 0u, 0u, 0u);

    // ones B-fragment (bf16 1.0 everywhere) for row-sum WMMA
    FragAB ones;
    #pragma unroll
    for (int e = 0; e < 16; ++e) ones.u[e] = 0x3F80;

    v8f o    = {};   // output accumulator (16x16)
    v8f lacc = {};   // softmax denominator accumulator (all cols identical)
    float m_run[8];
    #pragma unroll
    for (int i = 0; i < 8; ++i) m_run[i] = -3.0e38f;

    // fold 1/sqrt(HD) and log2(e) into one score scale: exp(x/4) == exp2(x*kscale)
    const float kscale = 0.25f * 1.44269504088896340736f;

    for (int ckv = 0; ckv < SKV_; ckv += 64) {
        __syncthreads();
        // ---- cooperative K/V chunk staging ----
        if (t < 128) {
            int r = t >> 1, half = t & 1;
            const unsigned short* gp = KVb +
                (size_t)(b * SKV_ + ckv + r) * (2 * DR_) + h * HD_ + 8 * half;
#if USE_ASYNC_LDS
            __builtin_amdgcn_global_load_async_to_lds_b128(
                (GAS_ v4i_*)gp, (LAS_ v4i_*)&Ks[r][8 * half], 0, 0);
#else
            *(uint4*)&Ks[r][8 * half] = *(const uint4*)gp;
#endif
            if (ckv + 64 < SKV_)
                __builtin_prefetch(gp + (size_t)64 * (2 * DR_), 0, 3);
        } else {
            int u = t - 128;
            int r = u >> 1, half = u & 1;
            const unsigned short* gp = KVb +
                (size_t)(b * SKV_ + ckv + r) * (2 * DR_) + DR_ + h * HD_ + 8 * half;
            union { uint4 q; unsigned short s[8]; } vv;
            vv.q = *(const uint4*)gp;
            #pragma unroll
            for (int j = 0; j < 8; ++j) Vt[8 * half + j][r] = vv.s[j];
            if (ckv + 64 < SKV_)
                __builtin_prefetch(gp + (size_t)64 * (2 * DR_), 0, 3);
        }
#if USE_ASYNC_LDS
        __builtin_amdgcn_s_wait_asynccnt(0);
#endif
        __syncthreads();

        // ---- scores: 4 x WMMA (16x16, K=16 zero-padded to 32) ----
        v8f s[4];
        #pragma unroll
        for (int nf = 0; nf < 4; ++nf) {
            FragAB bk;
            bk.q[0] = *(const uint4*)&Ks[nf * 16 + ln][d0];
            bk.q[1] = make_uint4(0u, 0u, 0u, 0u);
            v8f z = {};
            s[nf] = __builtin_amdgcn_wmma_f32_16x16x32_bf16(
                false, aq.v, false, bk.v, (short)0, z, false, false);
        }
        #pragma unroll
        for (int nf = 0; nf < 4; ++nf)
            #pragma unroll
            for (int i = 0; i < 8; ++i) s[nf][i] *= kscale;

        // ---- online softmax (exp2 domain): per-row max over 64 cols ----
        float alpha[8];
        #pragma unroll
        for (int i = 0; i < 8; ++i) {
            float v0 = fmaxf(fmaxf(s[0][i], s[1][i]), fmaxf(s[2][i], s[3][i]));
            #pragma unroll
            for (int off = 1; off < 16; off <<= 1)
                v0 = fmaxf(v0, __shfl_xor(v0, off, 32));
            float mn = fmaxf(m_run[i], v0);
            alpha[i] = fast_exp2(m_run[i] - mn);
            m_run[i] = mn;
        }
        #pragma unroll
        for (int nf = 0; nf < 4; ++nf) {
            #pragma unroll
            for (int i = 0; i < 8; ++i) {
                float p = fast_exp2(s[nf][i] - m_run[i]);
                Ps[w][i + 8 * kh][nf * 16 + ln] = f2bf(p);  // C-layout -> LDS
            }
        }
        #pragma unroll
        for (int i = 0; i < 8; ++i) {
            o[i]    *= alpha[i];
            lacc[i] *= alpha[i];
        }

        // ---- O += P(16x64) * V(64x16); denom += P @ ones (two K=32 steps) ----
        #pragma unroll
        for (int tc = 0; tc < 2; ++tc) {
            FragAB pa, vb;
            pa.q[0] = *(const uint4*)&Ps[w][ln][tc * 32 + d0];
            pa.q[1] = *(const uint4*)&Ps[w][ln][tc * 32 + d0 + 16];
            vb.q[0] = *(const uint4*)&Vt[ln][tc * 32 + d0];
            vb.q[1] = *(const uint4*)&Vt[ln][tc * 32 + d0 + 16];
            o = __builtin_amdgcn_wmma_f32_16x16x32_bf16(
                false, pa.v, false, vb.v, (short)0, o, false, false);
            lacc = __builtin_amdgcn_wmma_f32_16x16x32_bf16(
                false, pa.v, false, ones.v, (short)0, lacc, false, false);
        }
    }

    // ---- normalize and store ----
    #pragma unroll
    for (int i = 0; i < 8; ++i) {
        float val = o[i] / lacc[i];
        int r = b * SQ_ + qblk * 128 + w * 16 + i + 8 * kh;
        Ob[(size_t)r * DR_ + h * HD_ + ln] = f2bf(val);
    }
}

// ---------------------------------------------------------------------------
extern "C" void kernel_launch(void* const* d_in, const int* in_sizes, int n_in,
                              void* d_out, int out_size, void* d_ws, size_t ws_size,
                              hipStream_t stream)
{
    const float* hs  = (const float*)d_in[0];  // (B,SQ,D)
    const float* ehs = (const float*)d_in[1];  // (B,SKV,D)
    const float* Wq  = (const float*)d_in[2];
    const float* bq  = (const float*)d_in[3];
    const float* Wkv = (const float*)d_in[4];
    const float* bkv = (const float*)d_in[5];
    const float* Wp  = (const float*)d_in[6];
    const float* bp  = (const float*)d_in[7];
    float* out = (float*)d_out;

    const int M = B_ * SQ_;  // 8192 rows (batch flattened for projections)

    unsigned short* Qb  = (unsigned short*)d_ws;           // 8192*192 bf16
    unsigned short* KVb = Qb  + (size_t)M * DR_;           // 8192*384 bf16
    unsigned short* Ob  = KVb + (size_t)M * 2 * DR_;       // 8192*192 bf16

    dim3 blk(256);
    // Q projection: 8192x768 @ 768x192
    gemm_bias_wmma_k<float, unsigned short>
        <<<dim3(DR_ / 64, M / 128), blk, 0, stream>>>(hs, Wq, bq, Qb, M, DR_, D_);
    // KV projection: 8192x768 @ 768x384
    gemm_bias_wmma_k<float, unsigned short>
        <<<dim3(2 * DR_ / 64, M / 128), blk, 0, stream>>>(ehs, Wkv, bkv, KVb, M, 2 * DR_, D_);
    // Attention: B*H*(SQ/128) blocks
    attn_wmma_k<<<dim3(SQ_ / 128, H_, B_), blk, 0, stream>>>(Qb, KVb, Ob);
    // Output projection: 8192x192 @ 192x768 -> fp32 out
    gemm_bias_wmma_k<unsigned short, float>
        <<<dim3(D_ / 64, M / 128), blk, 0, stream>>>(Ob, Wp, bp, out, M, D_, DR_);
}
